// GATLinkPredictor_77481210020189
// MI455X (gfx1250) — matrix operand order repaired and checked by reading, not compiled, not run on hardware
//
#include <hip/hip_runtime.h>
#include <hip/hip_bf16.h>

#define NN   100000
#define INF_ 128
#define HH   4
#define CC   64
#define HC   256
#define E0E  1600000
#define ETOT (E0E + NN)
#define NEG  0.2f

typedef __attribute__((ext_vector_type(16))) __bf16 v16bf;
typedef __attribute__((ext_vector_type(8)))  float  v8f;

union Frag { v16bf v; unsigned u[8]; };

__device__ __forceinline__ unsigned pack_bf16(float a, float b) {
  // round-to-nearest-even fp32 -> bf16, packed into one dword
  unsigned ua = __float_as_uint(a), ub = __float_as_uint(b);
  ua += 0x7FFFu + ((ua >> 16) & 1u);
  ub += 0x7FFFu + ((ub >> 16) & 1u);
  return (ua >> 16) | (ub & 0xFFFF0000u);
}

// ---------------------------------------------------------------------------
// Kernel 1: h[N,256] = x[N,128] @ W[128,256] via bf16 WMMA, fp32 accumulate.
// Block = 256 threads (8 waves). Block tile = 128 rows x 256 cols.
// W staged in LDS as packed bf16 K-pairs: sW[k2][col], 64 KB, conflict-free.
// Each wave: 16 rows, 16 col-tiles, K = 4 x 32  ->  64 v_wmma per wave.
// ---------------------------------------------------------------------------
__global__ __launch_bounds__(256, 1)
void gemm_h_kernel(const float* __restrict__ x, const float* __restrict__ W,
                   float* __restrict__ h) {
  __shared__ unsigned sW[64 * 256];          // [k2][col] packed bf16 pairs
  const int t = threadIdx.x;
  const long long blockR = (long long)blockIdx.x * 128;

  // Stage W: pair p -> col = p%256 (coalesced), k2 = p/256
  #pragma unroll 4
  for (int i = 0; i < 64; ++i) {
    int p   = t + i * 256;
    int col = p & 255;
    int k2  = p >> 8;
    float w0 = W[(2 * k2)     * 256 + col];
    float w1 = W[(2 * k2 + 1) * 256 + col];
    sW[k2 * 256 + col] = pack_bf16(w0, w1);
  }
  __syncthreads();

  const int wave = t >> 5, lane = t & 31;
  const int half = lane >> 4, l16 = lane & 15;

  long long rg = blockR + wave * 16 + l16;
  if (rg >= NN) rg = NN - 1;                 // clamp reads; stores are guarded
  const float2* xrow = (const float2*)(x + rg * INF_);

  // A fragments: contiguous k order (same as B) — lane<16: K0..15, lane>=16: K16..31
  Frag a[4];
  #pragma unroll
  for (int kk = 0; kk < 4; ++kk) {
    const int kb = kk * 32 + half * 16;
    #pragma unroll
    for (int v = 0; v < 8; ++v) {
      int k = kb + 2 * v;
      float2 f = xrow[k >> 1];
      a[kk].u[v] = pack_bf16(f.x, f.y);
    }
  }

  v8f acc[16];
  const v8f zero = {0.f, 0.f, 0.f, 0.f, 0.f, 0.f, 0.f, 0.f};
  #pragma unroll
  for (int n = 0; n < 16; ++n) acc[n] = zero;

  #pragma unroll
  for (int n = 0; n < 16; ++n) {
    const int col = n * 16 + l16;
    #pragma unroll
    for (int kk = 0; kk < 4; ++kk) {
      Frag b;                                 // B 32x16: lane<16 K0..15, lane>=16 K16..31
      const int k2b = kk * 16 + half * 8;
      #pragma unroll
      for (int v = 0; v < 8; ++v) b.u[v] = sW[(k2b + v) * 256 + col];
      acc[n] = __builtin_amdgcn_wmma_f32_16x16x32_bf16(
          false, a[kk].v, false, b.v, (short)0, acc[n], false, false);
    }
  }

  // D layout: col = lane%16, row = 8*(lane/16) + vgpr
  #pragma unroll
  for (int n = 0; n < 16; ++n) {
    const int col = n * 16 + l16;
    #pragma unroll
    for (int v = 0; v < 8; ++v) {
      long long row = blockR + wave * 16 + half * 8 + v;
      if (row < NN) h[row * HC + col] = acc[n][v];
    }
  }
}

// ---------------------------------------------------------------------------
// Kernel 2: a_src[n,h] = sum_c h[n,h,c]*att_src[h,c]; same for a_dst.
// ---------------------------------------------------------------------------
__global__ void att_scores_kernel(const float* __restrict__ h,
                                  const float* __restrict__ att_src,
                                  const float* __restrict__ att_dst,
                                  float* __restrict__ a_src,
                                  float* __restrict__ a_dst) {
  int nh = blockIdx.x * blockDim.x + threadIdx.x;
  if (nh >= NN * HH) return;
  int n = nh >> 2, hd = nh & 3;
  const float4* hp = (const float4*)(h + (long long)n * HC + hd * CC);
  const float4* sp = (const float4*)(att_src + hd * CC);
  const float4* dp = (const float4*)(att_dst + hd * CC);
  float ss = 0.f, dd = 0.f;
  #pragma unroll
  for (int i = 0; i < 16; ++i) {
    float4 hv = hp[i], sv = sp[i], dv = dp[i];
    ss += hv.x * sv.x + hv.y * sv.y + hv.z * sv.z + hv.w * sv.w;
    dd += hv.x * dv.x + hv.y * dv.y + hv.z * dv.z + hv.w * dv.w;
  }
  a_src[nh] = ss;
  a_dst[nh] = dd;
}

// ---------------------------------------------------------------------------
// Kernel 3: init m = -inf, z = 0, outacc = 0.
// ---------------------------------------------------------------------------
__global__ void init_kernel(float* __restrict__ m, float* __restrict__ z,
                            float* __restrict__ outacc) {
  long long i = (long long)blockIdx.x * blockDim.x + threadIdx.x;
  if (i < (long long)NN * HH) { m[i] = -__builtin_huge_valf(); z[i] = 0.f; }
  if (i < (long long)NN * HC) outacc[i] = 0.f;
}

__device__ __forceinline__ void edge_sd(const long long* __restrict__ ei,
                                        long long e, int& s, int& d) {
  if (e < E0E) { s = (int)ei[e]; d = (int)ei[E0E + e]; }
  else         { s = d = (int)(e - E0E); }          // self loops
}

__device__ __forceinline__ float leaky(float v) {
  return (v > 0.f) ? v : v * NEG;
}

// ---------------------------------------------------------------------------
// Kernel 4: segment-max of leaky_relu(a_src[src]+a_dst[dst]) over dst.
// ---------------------------------------------------------------------------
__global__ void edge_max_kernel(const long long* __restrict__ ei,
                                const float* __restrict__ a_src,
                                const float* __restrict__ a_dst,
                                float* __restrict__ m) {
  long long idx = (long long)blockIdx.x * blockDim.x + threadIdx.x;
  if (idx >= (long long)ETOT * HH) return;
  int hd = (int)(idx & 3);
  long long e = idx >> 2;
  int s, d; edge_sd(ei, e, s, d);
  float lg = leaky(a_src[s * HH + hd] + a_dst[d * HH + hd]);
  float* addr = &m[d * HH + hd];
  // float atomic max via monotone int mappings (init is -inf)
  if (lg >= 0.f) atomicMax((int*)addr, __float_as_int(lg));
  else           atomicMin((unsigned int*)addr, __float_as_uint(lg));
}

// ---------------------------------------------------------------------------
// Kernel 5: z[dst,h] += exp(logit - m[dst,h]).
// ---------------------------------------------------------------------------
__global__ void edge_sum_kernel(const long long* __restrict__ ei,
                                const float* __restrict__ a_src,
                                const float* __restrict__ a_dst,
                                const float* __restrict__ m,
                                float* __restrict__ z) {
  long long idx = (long long)blockIdx.x * blockDim.x + threadIdx.x;
  if (idx >= (long long)ETOT * HH) return;
  int hd = (int)(idx & 3);
  long long e = idx >> 2;
  int s, d; edge_sd(ei, e, s, d);
  float lg = leaky(a_src[s * HH + hd] + a_dst[d * HH + hd]);
  float ev = __expf(lg - m[d * HH + hd]);
  atomicAdd(&z[d * HH + hd], ev);
}

// ---------------------------------------------------------------------------
// Kernel 6: outacc[dst,h,c] += alpha * h[src,h,c]. One thread per (e,h,c);
// 64 consecutive threads share all scalars and touch 64 contiguous floats
// (coalesced gather + coalesced atomics).
// ---------------------------------------------------------------------------
__global__ void edge_scatter_kernel(const long long* __restrict__ ei,
                                    const float* __restrict__ a_src,
                                    const float* __restrict__ a_dst,
                                    const float* __restrict__ m,
                                    const float* __restrict__ z,
                                    const float* __restrict__ h,
                                    float* __restrict__ outacc) {
  long long idx = (long long)blockIdx.x * blockDim.x + threadIdx.x;
  if (idx >= (long long)ETOT * HC) return;
  int c  = (int)(idx & 63);
  int hd = (int)((idx >> 6) & 3);
  long long e = idx >> 8;
  int s, d; edge_sd(ei, e, s, d);
  float lg = leaky(a_src[s * HH + hd] + a_dst[d * HH + hd]);
  float alpha = __expf(lg - m[d * HH + hd]) / (z[d * HH + hd] + 1e-16f);
  float val = h[(long long)s * HC + hd * CC + c] * alpha;
  atomicAdd(&outacc[(long long)d * HC + hd * CC + c], val);
}

// ---------------------------------------------------------------------------
// Kernel 7: head-mean + bias -> relu -> dot fc_w + fc_b -> sigmoid.
// One wave32 per node (2 channels/lane), shuffle reduction.
// ---------------------------------------------------------------------------
__global__ __launch_bounds__(256)
void finalize_kernel(const float* __restrict__ outacc,
                     const float* __restrict__ bias,
                     const float* __restrict__ fc_w,
                     const float* __restrict__ fc_b,
                     float* __restrict__ out) {
  int wave = threadIdx.x >> 5, lane = threadIdx.x & 31;
  long long n = (long long)blockIdx.x * 8 + wave;
  if (n >= NN) return;
  float partial = 0.f;
  #pragma unroll
  for (int j = 0; j < 2; ++j) {
    int c = lane * 2 + j;
    float s = 0.f;
    #pragma unroll
    for (int hd = 0; hd < HH; ++hd) s += outacc[n * HC + hd * CC + c];
    s = s * 0.25f + bias[c];
    s = fmaxf(s, 0.f);
    partial += s * fc_w[c];
  }
  #pragma unroll
  for (int off = 16; off > 0; off >>= 1)
    partial += __shfl_down(partial, off, 32);
  if (lane == 0) out[n] = 1.f / (1.f + __expf(-(partial + fc_b[0])));
}

// ---------------------------------------------------------------------------
extern "C" void kernel_launch(void* const* d_in, const int* in_sizes, int n_in,
                              void* d_out, int out_size, void* d_ws, size_t ws_size,
                              hipStream_t stream) {
  (void)in_sizes; (void)n_in; (void)out_size; (void)ws_size;
  const float*     x       = (const float*)d_in[0];
  const long long* ei      = (const long long*)d_in[1];   // int64 edge_index [2,E0]
  const float*     W       = (const float*)d_in[2];
  const float*     att_src = (const float*)d_in[3];
  const float*     att_dst = (const float*)d_in[4];
  const float*     bias    = (const float*)d_in[5];
  const float*     fc_w    = (const float*)d_in[6];
  const float*     fc_b    = (const float*)d_in[7];
  float* out = (float*)d_out;

  float* ws     = (float*)d_ws;
  float* h      = ws;                              // N*256
  float* a_src  = h + (long long)NN * HC;          // N*4
  float* a_dst  = a_src + (long long)NN * HH;      // N*4
  float* m      = a_dst + (long long)NN * HH;      // N*4
  float* z      = m + (long long)NN * HH;          // N*4
  float* outacc = z + (long long)NN * HH;          // N*256

  init_kernel<<<((long long)NN * HC + 255) / 256, 256, 0, stream>>>(m, z, outacc);
  gemm_h_kernel<<<(NN + 127) / 128, 256, 0, stream>>>(x, W, h);
  att_scores_kernel<<<(NN * HH + 255) / 256, 256, 0, stream>>>(h, att_src, att_dst,
                                                               a_src, a_dst);
  long long eh = (long long)ETOT * HH;
  edge_max_kernel<<<(unsigned)((eh + 255) / 256), 256, 0, stream>>>(ei, a_src, a_dst, m);
  edge_sum_kernel<<<(unsigned)((eh + 255) / 256), 256, 0, stream>>>(ei, a_src, a_dst, m, z);
  long long ec = (long long)ETOT * HC;
  edge_scatter_kernel<<<(unsigned)((ec + 255) / 256), 256, 0, stream>>>(ei, a_src, a_dst,
                                                                        m, z, h, outacc);
  finalize_kernel<<<(NN + 7) / 8, 256, 0, stream>>>(outacc, bias, fc_w, fc_b, out);
}
